// Attention_31808527794338
// MI455X (gfx1250) — matrix operand order repaired and checked by reading, not compiled, not run on hardware
//
#include <hip/hip_runtime.h>
#include <math.h>

// ---------------------------------------------------------------------------
// Problem constants (from reference)
// ---------------------------------------------------------------------------
constexpr int cB   = 2;
constexpr int cS   = 2048;
constexpr int cH   = 2048;
constexpr int cNH  = 16;
constexpr int cNKV = 4;
constexpr int cHD  = 128;
constexpr float cQSCALE = 0.08838834764831845f; // 1/sqrt(128)
constexpr float cNEG    = -3.0e38f;

// ---------------------------------------------------------------------------
// WMMA types / helpers (gfx1250, wave32)
// ---------------------------------------------------------------------------
typedef __attribute__((ext_vector_type(16))) __bf16 v16bf;
typedef __attribute__((ext_vector_type(8)))  __bf16 v8bf;
typedef __attribute__((ext_vector_type(8)))  float  v8f;

struct BFrag { v16bf hi; v16bf lo; };

static __device__ __forceinline__ void splitf(float f, __bf16& h, __bf16& l) {
    h = (__bf16)f;
    l = (__bf16)(f - (float)h);
}

static __device__ __forceinline__ v16bf cat8(v8bf a, v8bf b) {
    return __builtin_shufflevector(a, b, 0,1,2,3,4,5,6,7,8,9,10,11,12,13,14,15);
}

// D = A*B + C with f32 operands split into bf16 hi/lo (3 WMMAs, ~fp32 accuracy)
static __device__ __forceinline__ v8f mma3(const BFrag& a, const BFrag& b, v8f c) {
    c = __builtin_amdgcn_wmma_f32_16x16x32_bf16(false, a.hi, false, b.hi, (short)0, c, false, false);
    c = __builtin_amdgcn_wmma_f32_16x16x32_bf16(false, a.hi, false, b.lo, (short)0, c, false, false);
    c = __builtin_amdgcn_wmma_f32_16x16x32_bf16(false, a.lo, false, b.hi, (short)0, c, false, false);
    return c;
}

// A-fragment (16x32) from row-major bf16 hi/lo planes; contiguous b128 loads.
// ISA layout: lanes 0-15 row M=lane, slots 0..7 = K k0..k0+7, slots 8..15 = K+16..+23;
//             lanes 16-31 same rows, K offsets +8 / +24.
static __device__ __forceinline__ BFrag load_a_bf(const __bf16* __restrict__ hi,
                                                  const __bf16* __restrict__ lo,
                                                  int lda, int m0, int k0, int lane) {
    const int half = lane >> 4;
    const int m    = m0 + (lane & 15);
    const int kb   = k0 + half * 8;
    const __bf16* ph = hi + (size_t)m * lda + kb;
    const __bf16* pl = lo + (size_t)m * lda + kb;
    BFrag f;
    f.hi = cat8(*(const v8bf*)ph, *(const v8bf*)(ph + 16));
    f.lo = cat8(*(const v8bf*)pl, *(const v8bf*)(pl + 16));
    return f;
}

// B-fragment (32x16): contraction dim contiguous in memory (transposed operand).
// value(k, n) at base[n*ldk + k]. Lanes 0-15: col n, K k0..k0+7, k0+16..+23; lanes 16-31: K+8.
static __device__ __forceinline__ BFrag load_b_bf(const __bf16* __restrict__ hi,
                                                  const __bf16* __restrict__ lo,
                                                  int ldk, int n0, int k0, int lane) {
    const int half = lane >> 4;
    const int n    = n0 + (lane & 15);
    const int kb   = k0 + half * 8;
    const __bf16* ph = hi + (size_t)n * ldk + kb;
    const __bf16* pl = lo + (size_t)n * ldk + kb;
    BFrag f;
    f.hi = cat8(*(const v8bf*)ph, *(const v8bf*)(ph + 16));
    f.lo = cat8(*(const v8bf*)pl, *(const v8bf*)(pl + 16));
    return f;
}

// ---------------------------------------------------------------------------
// Conversion pass 1: elementwise f32 -> bf16 hi/lo planes (8 elems/thread)
// ---------------------------------------------------------------------------
__global__ __launch_bounds__(256) void split_kernel(const float* __restrict__ in,
                                                    __bf16* __restrict__ hi,
                                                    __bf16* __restrict__ lo) {
    const size_t i = ((size_t)blockIdx.x * blockDim.x + threadIdx.x) * 8;
    const float4 u0 = ((const float4*)(in + i))[0];
    const float4 u1 = ((const float4*)(in + i))[1];
    float v[8] = {u0.x, u0.y, u0.z, u0.w, u1.x, u1.y, u1.z, u1.w};
    v8bf h, l;
#pragma unroll
    for (int e = 0; e < 8; ++e) { __bf16 a, b; splitf(v[e], a, b); h[e] = a; l[e] = b; }
    *(v8bf*)(hi + i) = h;
    *(v8bf*)(lo + i) = l;
}

// ---------------------------------------------------------------------------
// Conversion pass 2: transpose + split. in: K x N row-major f32.
// out planes: N x K row-major bf16. 32x32 LDS tile.
// ---------------------------------------------------------------------------
__global__ __launch_bounds__(256) void tsplit_kernel(const float* __restrict__ in,
                                                     __bf16* __restrict__ hi,
                                                     __bf16* __restrict__ lo,
                                                     int K, int N) {
    __shared__ float tile[32][33];
    const int tx = threadIdx.x, ty = threadIdx.y;
    const int n0 = blockIdx.x * 32, k0 = blockIdx.y * 32;
#pragma unroll
    for (int r = 0; r < 4; ++r)
        tile[ty + 8 * r][tx] = in[(size_t)(k0 + ty + 8 * r) * N + n0 + tx];
    __syncthreads();
#pragma unroll
    for (int r = 0; r < 4; ++r) {
        const float v = tile[tx][ty + 8 * r];
        __bf16 h, l; splitf(v, h, l);
        const size_t o = (size_t)(n0 + ty + 8 * r) * K + k0 + tx;
        hi[o] = h; lo[o] = l;
    }
}

// ---------------------------------------------------------------------------
// Kernel 1: fused QKV projection + RoPE. One wave -> 32 seq rows x 128 head dims.
// Strip hs: [0,16)=Q head, [16,20)=K head, [20,24)=V head.
// Outputs bf16 hi/lo planes: Q,K row-major (S x HD); V transposed (HD x S).
// ---------------------------------------------------------------------------
__global__ __launch_bounds__(128) void qkv_rope_kernel(
    const __bf16* __restrict__ xhi,  const __bf16* __restrict__ xlo,
    const float*  __restrict__ cosT, const float*  __restrict__ sinT,
    const __bf16* __restrict__ wqhi, const __bf16* __restrict__ wqlo,
    const __bf16* __restrict__ wkhi, const __bf16* __restrict__ wklo,
    __bf16* __restrict__ qhi, __bf16* __restrict__ qlo,
    __bf16* __restrict__ khi, __bf16* __restrict__ klo,
    __bf16* __restrict__ vhi, __bf16* __restrict__ vlo) {

    const int lane = threadIdx.x & 31;
    const int unit = blockIdx.x * 4 + (threadIdx.x >> 5);
    const int stripsPerB = (cS / 32) * 24;
    const int b   = unit / stripsPerB;
    const int rem = unit % stripsPerB;
    const int st  = rem / 24;       // 32-row seq tile
    const int hs  = rem % 24;       // head strip
    const int s0  = st * 32;

    const __bf16 *Whi, *Wlo; int wcol; bool rope, isV; float scale;
    __bf16 *dhi, *dlo;
    if (hs < 16) {
        Whi = wqhi; Wlo = wqlo; wcol = hs * cHD;
        dhi = qhi + (size_t)(b * cNH + hs) * cS * cHD;
        dlo = qlo + (size_t)(b * cNH + hs) * cS * cHD;
        rope = true; isV = false; scale = cQSCALE;
    } else if (hs < 20) {
        const int kh = hs - 16;
        Whi = wkhi; Wlo = wklo; wcol = kh * cHD;
        dhi = khi + (size_t)(b * cNKV + kh) * cS * cHD;
        dlo = klo + (size_t)(b * cNKV + kh) * cS * cHD;
        rope = true; isV = false; scale = 1.0f;
    } else {
        const int vh = hs - 20;
        Whi = wkhi; Wlo = wklo; wcol = cNKV * cHD + vh * cHD;
        dhi = vhi + (size_t)(b * cNKV + vh) * cHD * cS;   // transposed dest
        dlo = vlo + (size_t)(b * cNKV + vh) * cHD * cS;
        rope = false; isV = true; scale = 1.0f;
    }

    const __bf16* xbh = xhi + (size_t)b * cS * cH;
    const __bf16* xbl = xlo + (size_t)b * cS * cH;

    v8f c[2][8];
#pragma unroll
    for (int mi = 0; mi < 2; ++mi)
#pragma unroll
        for (int t = 0; t < 8; ++t) { v8f z = {}; c[mi][t] = z; }

#pragma unroll 1
    for (int k0 = 0; k0 < cH; k0 += 32) {
        BFrag a0 = load_a_bf(xbh, xbl, cH, s0,      k0, lane);
        BFrag a1 = load_a_bf(xbh, xbl, cH, s0 + 16, k0, lane);
#pragma unroll
        for (int t = 0; t < 8; ++t) {
            BFrag bb = load_b_bf(Whi, Wlo, cH, wcol + 16 * t, k0, lane);
            c[0][t] = mma3(a0, bb, c[0][t]);
            c[1][t] = mma3(a1, bb, c[1][t]);
        }
    }

    // Epilogue: RoPE (tile t pairs with t^4: rotate_half pairs d <-> d +/- 64)
    const int half = lane >> 4;
    const int nl   = lane & 15;
#pragma unroll
    for (int mi = 0; mi < 2; ++mi)
#pragma unroll
    for (int t = 0; t < 8; ++t)
#pragma unroll
    for (int r = 0; r < 8; ++r) {
        const int srow = s0 + mi * 16 + r + 8 * half;
        const int j    = t * 16 + nl;
        float v = c[mi][t][r];
        if (rope) {
            const float cv = cosT[(size_t)srow * cHD + j];
            const float sv = sinT[(size_t)srow * cHD + j];
            const float pr = c[mi][t ^ 4][r];
            v = v * cv + (t < 4 ? -pr : pr) * sv;
        }
        v *= scale;
        __bf16 h, l; splitf(v, h, l);
        const size_t o = isV ? ((size_t)j * cS + srow) : ((size_t)srow * cHD + j);
        dhi[o] = h; dlo[o] = l;
    }
}

// ---------------------------------------------------------------------------
// Kernel 2: causal flash attention (GQA). One wave per (b, h, 16-row q tile).
// Single-wave workgroups -> __syncthreads is safe in the causally-divergent loop.
// ---------------------------------------------------------------------------
__global__ __launch_bounds__(32) void attn_kernel(
    const __bf16* __restrict__ qhi, const __bf16* __restrict__ qlo,
    const __bf16* __restrict__ khi, const __bf16* __restrict__ klo,
    const __bf16* __restrict__ vhi, const __bf16* __restrict__ vlo,
    __bf16* __restrict__ aohi, __bf16* __restrict__ aolo) {

    const int lane = threadIdx.x & 31;
    const int qtPerH = cS / 16;
    const int unit = blockIdx.x;
    const int qt = unit % qtPerH;
    const int h  = (unit / qtPerH) % cNH;
    const int b  = unit / (qtPerH * cNH);
    const int q0 = qt * 16;
    const int hk = h / (cNH / cNKV);

    const __bf16* Qh  = qhi + (size_t)(b * cNH  + h ) * cS * cHD;
    const __bf16* Ql  = qlo + (size_t)(b * cNH  + h ) * cS * cHD;
    const __bf16* Kh  = khi + (size_t)(b * cNKV + hk) * cS * cHD;
    const __bf16* Kl  = klo + (size_t)(b * cNKV + hk) * cS * cHD;
    const __bf16* Vth = vhi + (size_t)(b * cNKV + hk) * cHD * cS;  // HD x S
    const __bf16* Vtl = vlo + (size_t)(b * cNKV + hk) * cHD * cS;

    // Preload Q fragments over full head dim (4 k-chunks of 32)
    BFrag qf[4];
#pragma unroll
    for (int dk = 0; dk < 4; ++dk) qf[dk] = load_a_bf(Qh, Ql, cHD, q0, dk * 32, lane);

    v8f o[8];
#pragma unroll
    for (int t = 0; t < 8; ++t) { v8f z = {}; o[t] = z; }
    float m[8], l[8];
#pragma unroll
    for (int r = 0; r < 8; ++r) { m[r] = cNEG; l[r] = 0.0f; }

    __shared__ __bf16 sPhi[16 * 32];
    __shared__ __bf16 sPlo[16 * 32];

    const int half = lane >> 4;
    const int nl   = lane & 15;
    const int nkt  = (q0 + 16 + 31) >> 5;

#pragma unroll 1
    for (int kt = 0; kt < nkt; ++kt) {
        const int kv0 = kt * 32;

        // --- scores: two 16x16 tiles of (Q*scale) @ K^T ---
        v8f s0t = {}, s1t = {};
#pragma unroll
        for (int dk = 0; dk < 4; ++dk) {
            BFrag kb0 = load_b_bf(Kh, Kl, cHD, kv0, dk * 32, lane);
            s0t = mma3(qf[dk], kb0, s0t);
        }
#pragma unroll
        for (int dk = 0; dk < 4; ++dk) {
            BFrag kb1 = load_b_bf(Kh, Kl, cHD, kv0 + 16, dk * 32, lane);
            s1t = mma3(qf[dk], kb1, s1t);
        }

        // --- causal mask + online softmax (row = q0 + r + 8*half) ---
#pragma unroll
        for (int r = 0; r < 8; ++r) {
            const int row  = q0 + r + 8 * half;
            const int col0 = kv0 + nl;
            const int col1 = col0 + 16;
            float v0 = (col0 > row) ? cNEG : s0t[r];
            float v1 = (col1 > row) ? cNEG : s1t[r];

            float mx = fmaxf(v0, v1);
#pragma unroll
            for (int off = 8; off >= 1; off >>= 1)
                mx = fmaxf(mx, __shfl_xor(mx, off, 16));

            const float mn = fmaxf(m[r], mx);
            const float al = __expf(m[r] - mn);
            const float p0 = __expf(v0 - mn);
            const float p1 = __expf(v1 - mn);

            float rs = p0 + p1;
#pragma unroll
            for (int off = 8; off >= 1; off >>= 1)
                rs += __shfl_xor(rs, off, 16);

            l[r] = l[r] * al + rs;
            m[r] = mn;
#pragma unroll
            for (int t = 0; t < 8; ++t) o[t][r] *= al;

            const int rloc = r + 8 * half;
            __bf16 h0, l0, h1, l1;
            splitf(p0, h0, l0); splitf(p1, h1, l1);
            sPhi[rloc * 32 + nl]      = h0; sPlo[rloc * 32 + nl]      = l0;
            sPhi[rloc * 32 + 16 + nl] = h1; sPlo[rloc * 32 + 16 + nl] = l1;
        }

        __syncthreads();
        BFrag pf = load_a_bf(sPhi, sPlo, 32, 0, 0, lane);  // P in A-frag layout
        __syncthreads();

        // --- O += P @ V  (V transposed: contiguous along kv) ---
#pragma unroll
        for (int t = 0; t < 8; ++t) {
            BFrag vb = load_b_bf(Vth, Vtl, cS, t * 16, kv0, lane);
            o[t] = mma3(pf, vb, o[t]);
        }
    }

    // Epilogue: normalize, split, write AO hi/lo planes (B, S, NH*HD)
#pragma unroll
    for (int r = 0; r < 8; ++r) {
        const int   srow = q0 + r + 8 * half;
        const float inv  = 1.0f / l[r];
#pragma unroll
        for (int t = 0; t < 8; ++t) {
            const int j = t * 16 + nl;
            const float v = o[t][r] * inv;
            __bf16 h, lw; splitf(v, h, lw);
            const size_t off = ((size_t)(b * cS + srow)) * (cNH * cHD) + h * 0 + (size_t)::min(h,h); // placeholder avoided below
            (void)off;
            const size_t oo = ((size_t)(b * cS + srow)) * (cNH * cHD) + (size_t)h * cHD + j;
            aohi[oo] = h; aolo[oo] = lw;
        }
    }
}

// ---------------------------------------------------------------------------
// Kernel 3: output projection (B*S x 2048) @ Wo^T planes -> f32 out.
// One wave -> 32 rows x 128 cols.
// ---------------------------------------------------------------------------
__global__ __launch_bounds__(128) void oproj_kernel(
    const __bf16* __restrict__ ahi, const __bf16* __restrict__ alo,
    const __bf16* __restrict__ whi, const __bf16* __restrict__ wlo,
    float* __restrict__ Out, int M, int K, int N) {

    const int lane = threadIdx.x & 31;
    const int unit = blockIdx.x * 4 + (threadIdx.x >> 5);
    const int nstrips = N / 128;
    const int mt = unit / nstrips;
    const int ns = unit % nstrips;
    const int m0 = mt * 32;
    const int n0 = ns * 128;

    v8f c[2][8];
#pragma unroll
    for (int mi = 0; mi < 2; ++mi)
#pragma unroll
        for (int t = 0; t < 8; ++t) { v8f z = {}; c[mi][t] = z; }

#pragma unroll 1
    for (int k0 = 0; k0 < K; k0 += 32) {
        BFrag a0 = load_a_bf(ahi, alo, K, m0,      k0, lane);
        BFrag a1 = load_a_bf(ahi, alo, K, m0 + 16, k0, lane);
#pragma unroll
        for (int t = 0; t < 8; ++t) {
            BFrag bb = load_b_bf(whi, wlo, K, n0 + 16 * t, k0, lane);
            c[0][t] = mma3(a0, bb, c[0][t]);
            c[1][t] = mma3(a1, bb, c[1][t]);
        }
    }

    const int half = lane >> 4;
    const int nl   = lane & 15;
#pragma unroll
    for (int mi = 0; mi < 2; ++mi)
#pragma unroll
    for (int t = 0; t < 8; ++t)
#pragma unroll
    for (int r = 0; r < 8; ++r)
        Out[(size_t)(m0 + mi * 16 + r + 8 * half) * N + n0 + 16 * t + nl] = c[mi][t][r];
}

// ---------------------------------------------------------------------------
// Launch
// ---------------------------------------------------------------------------
extern "C" void kernel_launch(void* const* d_in, const int* in_sizes, int n_in,
                              void* d_out, int out_size, void* d_ws, size_t ws_size,
                              hipStream_t stream) {
    const float* x    = (const float*)d_in[0];
    const float* cosT = (const float*)d_in[1];
    const float* sinT = (const float*)d_in[2];
    const float* Wq   = (const float*)d_in[3];
    const float* Wkv  = (const float*)d_in[4];
    const float* Wo   = (const float*)d_in[5];
    float* out = (float*)d_out;

    __bf16* ws = (__bf16*)d_ws;
    const size_t nX  = (size_t)cB * cS * cH;          // 8,388,608
    const size_t nWq = (size_t)cH * cNH * cHD;        // 4,194,304
    const size_t nWk = (size_t)cH * 2 * cNKV * cHD;   // 2,097,152
    const size_t nWo = (size_t)cNH * cHD * cH;        // 4,194,304
    const size_t nQ  = (size_t)cB * cNH * cS * cHD;   // 8,388,608
    const size_t nKV = (size_t)cB * cNKV * cS * cHD;  // 2,097,152

    __bf16* p = ws;
    __bf16 *xhi = p, *xlo;   p += nX;  xlo = p; p += nX;
    __bf16 *wqhi = p, *wqlo; p += nWq; wqlo = p; p += nWq;   // transposed N x K
    __bf16 *wkhi = p, *wklo; p += nWk; wklo = p; p += nWk;   // transposed N x K
    __bf16 *wohi = p, *wolo; p += nWo; wolo = p; p += nWo;   // transposed N x K
    __bf16 *qhi = p, *qlo;   p += nQ;  qlo = p; p += nQ;
    __bf16 *khi = p, *klo;   p += nKV; klo = p; p += nKV;
    __bf16 *vhi = p, *vlo;   p += nKV; vlo = p; p += nKV;    // transposed HD x S
    __bf16 *aohi = p, *aolo; p += nQ;  aolo = p; p += nQ;

    // 0) Precision-split / transpose passes (bandwidth-bound, L2-resident after)
    split_kernel<<<(int)(nX / 8 / 256), 256, 0, stream>>>(x, xhi, xlo);
    tsplit_kernel<<<dim3(cNH * cHD / 32, cH / 32), dim3(32, 8), 0, stream>>>(Wq, wqhi, wqlo, cH, cNH * cHD);
    tsplit_kernel<<<dim3(2 * cNKV * cHD / 32, cH / 32), dim3(32, 8), 0, stream>>>(Wkv, wkhi, wklo, cH, 2 * cNKV * cHD);
    tsplit_kernel<<<dim3(cH / 32, cNH * cHD / 32), dim3(32, 8), 0, stream>>>(Wo, wohi, wolo, cNH * cHD, cH);

    // 1) QKV projection + RoPE: B*(S/32)*24 waves, 4 waves/block
    {
        const int units = cB * (cS / 32) * 24;            // 3072
        qkv_rope_kernel<<<units / 4, 128, 0, stream>>>(xhi, xlo, cosT, sinT,
                                                       wqhi, wqlo, wkhi, wklo,
                                                       qhi, qlo, khi, klo, vhi, vlo);
    }
    // 2) Flash attention: one wave per (b, h, q-tile)
    {
        const int units = cB * cNH * (cS / 16);           // 4096
        attn_kernel<<<units, 32, 0, stream>>>(qhi, qlo, khi, klo, vhi, vlo, aohi, aolo);
    }
    // 3) Output projection: (B*S x 2048) @ (2048 x 2048)
    {
        const int M = cB * cS, K = cNH * cHD, N = cH;
        const int units = (M / 32) * (N / 128);           // 2048
        oproj_kernel<<<units / 4, 128, 0, stream>>>(aohi, aolo, wohi, wolo, out, M, K, N);
    }
}
